// STLSTMClassifier_9251359555919
// MI455X (gfx1250) — matrix operand order repaired and checked by reading, not compiled, not run on hardware
//
#include <hip/hip_runtime.h>
#include <hip/hip_bf16.h>
#include <math.h>

// ---------------------------------------------------------------------------
// ST-LSTM classifier for MI455X (gfx1250), wave32 + WMMA bf16.
//   Phase 0: weight prep -> bf16 (W_cat = [w_ih | w_s | w_q] with zeroed g-gate
//            blocks), bias = b_ih + b_hh, zero h0/c0.
//   Phase 1: X_cat (t*B+b, 1536) bf16 = [batch_l | slot_interp_s | slot_interp_q]
//   Phase 2: P (T,B,4096) bf16 = X_cat @ W_cat^T + bias   (one big WMMA GEMM,
//            32Mx64N wave tile: every B-fragment feeds 2 WMMAs, A reused 4x)
//   Phase 3: 200x fused step kernel: gates = P[t] + h @ Whh^T (WMMA, C-init
//            from P), sigmoid/tanh epilogue, c in-place, h double-buffered bf16
//   Phase 4: out = hT @ lin_w^T + lin_b (WMMA, f32 out)
// ---------------------------------------------------------------------------

typedef __attribute__((ext_vector_type(16))) __bf16        v16bf;
typedef __attribute__((ext_vector_type(8)))  float         v8f;
typedef __attribute__((ext_vector_type(4)))  unsigned int  u32x4;

#define NSLOT   16
#define INSZ    512
#define HID     1024
#define OUTSZ   512
#define BSZ     256
#define TST     200
#define KCAT    1536          // 3*INSZ
#define G4      4096          // 4*HID

union Frag16 { u32x4 q[2]; v16bf v; };

__device__ __forceinline__ unsigned short f2bf(float f) {
    unsigned int u = __float_as_uint(f);
    u += 0x7FFFu + ((u >> 16) & 1u);          // round-to-nearest-even
    return (unsigned short)(u >> 16);
}
__device__ __forceinline__ float bf2f(unsigned short s) {
    return __uint_as_float(((unsigned int)s) << 16);
}
__device__ __forceinline__ float sigmoidf(float x) {
    return 1.0f / (1.0f + __expf(-x));
}

// A-fragment: 16x32 bf16, row = m0 + (lane&15), K chunks {0,16}/{8,24} by half.
__device__ __forceinline__ Frag16 load_afrag(const unsigned short* rowp, int k0, int hl) {
    Frag16 f;
    const unsigned short* ap = rowp + k0 + hl * 8;
    f.q[0] = *(const u32x4*)(ap);
    f.q[1] = *(const u32x4*)(ap + 16);
    return f;
}
// B-fragment: 32x16 bf16 from row-major (N x K) weights; K chunks {0,16}/{8,24} by half.
__device__ __forceinline__ Frag16 load_bfrag(const unsigned short* W, int n0, int lm,
                                             int ldk, int k0, int hl) {
    Frag16 f;
    const unsigned short* bp = W + (size_t)(n0 + lm) * ldk + k0 + hl * 8;
    f.q[0] = *(const u32x4*)(bp);
    f.q[1] = *(const u32x4*)(bp + 16);
    return f;
}

// ---------------- phase 0: weight / state prep ----------------

__global__ void k_build_wcat(const float* __restrict__ w_ih,
                             const float* __restrict__ w_s,
                             const float* __restrict__ w_q,
                             unsigned short* __restrict__ wcat) {
    int idx = blockIdx.x * blockDim.x + threadIdx.x;
    if (idx >= G4 * KCAT) return;
    int n = idx / KCAT, c = idx % KCAT;
    int gate = n >> 10, j = n & (HID - 1);
    float v;
    if (c < INSZ) {
        v = w_ih[n * INSZ + c];
    } else {
        int sr = (gate == 0) ? j : (gate == 1) ? (HID + j) : (gate == 3) ? (2 * HID + j) : -1;
        if (sr < 0) v = 0.0f;                       // g-gate gets no s/q contribution
        else if (c < 2 * INSZ) v = w_s[sr * INSZ + (c - INSZ)];
        else                   v = w_q[sr * INSZ + (c - 2 * INSZ)];
    }
    wcat[idx] = f2bf(v);
}

__global__ void k_cvt_bf16(const float* __restrict__ src,
                           unsigned short* __restrict__ dst, int n) {
    int idx = blockIdx.x * blockDim.x + threadIdx.x;
    if (idx < n) dst[idx] = f2bf(src[idx]);
}

__global__ void k_bias_init(const float* __restrict__ b_ih,
                            const float* __restrict__ b_hh,
                            float* __restrict__ bias) {
    int idx = blockIdx.x * blockDim.x + threadIdx.x;
    if (idx < G4) bias[idx] = b_ih[idx] + b_hh[idx];
}

__global__ void k_init_state(unsigned short* __restrict__ h0,
                             float* __restrict__ c0) {
    int idx = blockIdx.x * blockDim.x + threadIdx.x;
    if (idx < BSZ * HID) { h0[idx] = 0; c0[idx] = 0.0f; }
}

// ---------------- phase 1: X_cat with fused slot interpolation ----------------

__global__ void k_build_xcat(const float* __restrict__ batch_l,
                             const float* __restrict__ batch_t,
                             const float* __restrict__ batch_d,
                             const float* __restrict__ emb_s,
                             const float* __restrict__ emb_q,
                             unsigned short* __restrict__ xcat) {
    int r   = blockIdx.x;                                   // row = t*B + b
    int col = blockIdx.y * blockDim.x + threadIdx.x;        // 0..511
    int b = r % BSZ, t = r / BSZ;
    float tv = batch_t[b * TST + t];
    float dv = batch_d[b * TST + t];
    int k0t = (int)floorf(tv); k0t = k0t < 0 ? 0 : (k0t > NSLOT - 2 ? NSLOT - 2 : k0t);
    int k0d = (int)floorf(dv); k0d = k0d < 0 ? 0 : (k0d > NSLOT - 2 ? NSLOT - 2 : k0d);
    float ft = tv - (float)k0t;
    float fd = dv - (float)k0d;
    float xl = batch_l[((size_t)b * TST + t) * INSZ + col];
    float sv = (1.0f - ft) * emb_s[k0t * INSZ + col] + ft * emb_s[(k0t + 1) * INSZ + col];
    float qv = (1.0f - fd) * emb_q[k0d * INSZ + col] + fd * emb_q[(k0d + 1) * INSZ + col];
    size_t base = (size_t)r * KCAT;
    xcat[base + col]            = f2bf(xl);
    xcat[base + INSZ + col]     = f2bf(sv);
    xcat[base + 2 * INSZ + col] = f2bf(qv);
}

// ---------------- phase 2: P = X_cat @ W_cat^T + bias ----------------
// grid = (G4/256, M/64), block = 256 (8 waves); wave tile = 32M x 64N:
// 2 A-frags reused 4x each, 4 B-frags reused 2x each -> 8 WMMAs / 12 loads.

__global__ __launch_bounds__(256) void k_gemm_p(const unsigned short* __restrict__ X,
                                                const unsigned short* __restrict__ W,
                                                const float* __restrict__ bias,
                                                unsigned short* __restrict__ P) {
    int wave = threadIdx.x >> 5, lane = threadIdx.x & 31;
    int hl = lane >> 4, lm = lane & 15;
    int m0 = blockIdx.y * 64 + (wave & 1) * 32;             // 2 M-positions of 32
    int nb = blockIdx.x * 256 + (wave >> 1) * 64;           // 4 N-positions of 64
    v8f acc0[4] = {};                                       // rows m0..m0+15
    v8f acc1[4] = {};                                       // rows m0+16..m0+31
    const unsigned short* arow0 = X + (size_t)(m0 + lm) * KCAT;
    const unsigned short* arow1 = X + (size_t)(m0 + 16 + lm) * KCAT;
    for (int k0 = 0; k0 < KCAT; k0 += 32) {
        __builtin_prefetch(arow0 + k0 + 32, 0, 0);          // global_prefetch_b8
        __builtin_prefetch(arow1 + k0 + 32, 0, 0);
        Frag16 fa0 = load_afrag(arow0, k0, hl);
        Frag16 fa1 = load_afrag(arow1, k0, hl);
#pragma unroll
        for (int tt = 0; tt < 4; ++tt) {
            Frag16 fb = load_bfrag(W, nb + tt * 16, lm, KCAT, k0, hl);
            acc0[tt] = __builtin_amdgcn_wmma_f32_16x16x32_bf16(
                false, fa0.v, false, fb.v, (short)0, acc0[tt], false, false);
            acc1[tt] = __builtin_amdgcn_wmma_f32_16x16x32_bf16(
                false, fa1.v, false, fb.v, (short)0, acc1[tt], false, false);
        }
    }
#pragma unroll
    for (int tt = 0; tt < 4; ++tt) {
        int col = nb + tt * 16 + lm;
        float bv = bias[col];
#pragma unroll
        for (int r = 0; r < 8; ++r) {
            int row0 = m0 + r + hl * 8;
            P[(size_t)row0 * G4 + col]        = f2bf(acc0[tt][r] + bv);
            P[(size_t)(row0 + 16) * G4 + col] = f2bf(acc1[tt][r] + bv);
        }
    }
}

// ---------------- phase 3: fused LSTM step ----------------
// grid = (HID/32, BSZ/32), block = 128 (4 waves); wave tile = 16x16, all 4 gates
// at the same (m,n) so the LSTM pointwise math fuses into the GEMM epilogue.

__global__ __launch_bounds__(128) void k_step(const unsigned short* __restrict__ P_t,
                                              const unsigned short* __restrict__ Whh,
                                              const unsigned short* __restrict__ h_in,
                                              unsigned short* __restrict__ h_out,
                                              float* __restrict__ c) {
    int wave = threadIdx.x >> 5, lane = threadIdx.x & 31;
    int hl = lane >> 4, lm = lane & 15;
    int m0 = blockIdx.y * 32 + (wave & 1) * 16;
    int n0 = blockIdx.x * 32 + (wave >> 1) * 16;            // within HID
    v8f acc[4];
#pragma unroll
    for (int g = 0; g < 4; ++g) {                            // C-init from P[t]
        int col = g * HID + n0 + lm;
#pragma unroll
        for (int r = 0; r < 8; ++r)
            acc[g][r] = bf2f(P_t[(size_t)(m0 + r + hl * 8) * G4 + col]);
    }
    const unsigned short* arow = h_in + (size_t)(m0 + lm) * HID;
    for (int k0 = 0; k0 < HID; k0 += 32) {
        __builtin_prefetch(arow + k0 + 32, 0, 0);
        Frag16 fa = load_afrag(arow, k0, hl);
#pragma unroll
        for (int g = 0; g < 4; ++g) {
            Frag16 fb = load_bfrag(Whh, g * HID + n0, lm, HID, k0, hl);
            acc[g] = __builtin_amdgcn_wmma_f32_16x16x32_bf16(
                false, fa.v, false, fb.v, (short)0, acc[g], false, false);
        }
    }
#pragma unroll
    for (int r = 0; r < 8; ++r) {                            // LSTM pointwise epilogue
        size_t idx = (size_t)(m0 + r + hl * 8) * HID + (n0 + lm);
        float iv = sigmoidf(acc[0][r]);
        float fv = sigmoidf(acc[1][r]);
        float gv = tanhf(acc[2][r]);
        float ov = sigmoidf(acc[3][r]);
        float cy = fv * c[idx] + iv * gv;
        c[idx] = cy;
        h_out[idx] = f2bf(ov * tanhf(cy));
    }
}

// ---------------- phase 4: out = hT @ lin_w^T + lin_b ----------------
// grid = (OUTSZ/32, BSZ/32), block = 128 (4 waves).

__global__ __launch_bounds__(128) void k_gemm_out(const unsigned short* __restrict__ h,
                                                  const unsigned short* __restrict__ Wl,
                                                  const float* __restrict__ lin_b,
                                                  float* __restrict__ out) {
    int wave = threadIdx.x >> 5, lane = threadIdx.x & 31;
    int hl = lane >> 4, lm = lane & 15;
    int m0 = blockIdx.y * 32 + (wave & 1) * 16;
    int n0 = blockIdx.x * 32 + (wave >> 1) * 16;
    v8f acc = {};
    const unsigned short* arow = h + (size_t)(m0 + lm) * HID;
    for (int k0 = 0; k0 < HID; k0 += 32) {
        Frag16 fa = load_afrag(arow, k0, hl);
        Frag16 fb = load_bfrag(Wl, n0, lm, HID, k0, hl);
        acc = __builtin_amdgcn_wmma_f32_16x16x32_bf16(
            false, fa.v, false, fb.v, (short)0, acc, false, false);
    }
    int col = n0 + lm;
    float bv = lin_b[col];
#pragma unroll
    for (int r = 0; r < 8; ++r)
        out[(size_t)(m0 + r + hl * 8) * OUTSZ + col] = acc[r] + bv;
}

// ---------------- host-side orchestration ----------------

extern "C" void kernel_launch(void* const* d_in, const int* in_sizes, int n_in,
                              void* d_out, int out_size, void* d_ws, size_t ws_size,
                              hipStream_t stream) {
    const float* batch_l = (const float*)d_in[0];
    const float* batch_t = (const float*)d_in[1];
    const float* batch_d = (const float*)d_in[2];
    const float* emb_s   = (const float*)d_in[3];
    const float* emb_q   = (const float*)d_in[4];
    const float* w_ih    = (const float*)d_in[5];
    const float* w_hh    = (const float*)d_in[6];
    const float* w_s     = (const float*)d_in[7];
    const float* w_q     = (const float*)d_in[8];
    const float* b_ih    = (const float*)d_in[9];
    const float* b_hh    = (const float*)d_in[10];
    const float* lin_w   = (const float*)d_in[11];
    const float* lin_b   = (const float*)d_in[12];
    float* out = (float*)d_out;

    char* ws = (char*)d_ws;
    size_t off = 0;
    auto carve = [&](size_t bytes) -> void* {
        void* p = ws + off;
        off = (off + bytes + 255) & ~(size_t)255;
        return p;
    };
    unsigned short* Xcat  = (unsigned short*)carve((size_t)BSZ * TST * KCAT * 2);
    unsigned short* Wcat  = (unsigned short*)carve((size_t)G4 * KCAT * 2);
    unsigned short* WhhB  = (unsigned short*)carve((size_t)G4 * HID * 2);
    unsigned short* WlinB = (unsigned short*)carve((size_t)OUTSZ * HID * 2);
    float*          bias  = (float*)carve((size_t)G4 * 4);
    unsigned short* P     = (unsigned short*)carve((size_t)BSZ * TST * G4 * 2);
    unsigned short* h0    = (unsigned short*)carve((size_t)BSZ * HID * 2);
    unsigned short* h1    = (unsigned short*)carve((size_t)BSZ * HID * 2);
    float*          cst   = (float*)carve((size_t)BSZ * HID * 4);

    // phase 0
    k_build_wcat<<<(G4 * KCAT + 255) / 256, 256, 0, stream>>>(w_ih, w_s, w_q, Wcat);
    k_cvt_bf16<<<(G4 * HID + 255) / 256, 256, 0, stream>>>(w_hh, WhhB, G4 * HID);
    k_cvt_bf16<<<(OUTSZ * HID + 255) / 256, 256, 0, stream>>>(lin_w, WlinB, OUTSZ * HID);
    k_bias_init<<<(G4 + 255) / 256, 256, 0, stream>>>(b_ih, b_hh, bias);
    k_init_state<<<(BSZ * HID + 255) / 256, 256, 0, stream>>>(h0, cst);

    // phase 1
    dim3 gx(BSZ * TST, INSZ / 256);
    k_build_xcat<<<gx, 256, 0, stream>>>(batch_l, batch_t, batch_d, emb_s, emb_q, Xcat);

    // phase 2
    dim3 gp(G4 / 256, (BSZ * TST) / 64);
    k_gemm_p<<<gp, 256, 0, stream>>>(Xcat, Wcat, bias, P);

    // phase 3: sequential recurrence, grid-wide sync via kernel boundaries
    unsigned short* hprev = h0;
    unsigned short* hnext = h1;
    dim3 gs(HID / 32, BSZ / 32);
    for (int t = 0; t < TST; ++t) {
        k_step<<<gs, 128, 0, stream>>>(P + (size_t)t * BSZ * G4, WhhB, hprev, hnext, cst);
        unsigned short* tmp = hprev; hprev = hnext; hnext = tmp;
    }

    // phase 4
    dim3 go(OUTSZ / 32, BSZ / 32);
    k_gemm_out<<<go, 128, 0, stream>>>(hprev, WlinB, lin_b, out);
}
